// DIN_22368189678378
// MI455X (gfx1250) — matrix operand order repaired and checked by reading, not compiled, not run on hardware
//
#include <hip/hip_runtime.h>
#include <hip/hip_bf16.h>

typedef __attribute__((ext_vector_type(16))) _Float16 v16h;
typedef __attribute__((ext_vector_type(8)))  _Float16 v8h;
typedef __attribute__((ext_vector_type(4)))  _Float16 v4h;
typedef __attribute__((ext_vector_type(8)))  float    v8f;
typedef __attribute__((ext_vector_type(4)))  float    v4f;
typedef __attribute__((ext_vector_type(4)))  int      v4i;

#define BATCH 8192
#define LSEQ  200
#define LPAD  208       // 13 tiles of 16 rows (rows 200..207 zero padded)
#define DDIM  64
#define HS    72        // LDS row stride in halfs (144B: mult of 16B, conflict-friendly)
#define BN_EPS 1e-5f

#if __has_builtin(__builtin_amdgcn_global_load_async_to_lds_b128) && \
    __has_builtin(__builtin_amdgcn_s_wait_asynccnt)
#define HAS_ASYNC_LDS 1
typedef __attribute__((address_space(1))) v4i* gv4i_p;
typedef __attribute__((address_space(3))) v4i* lv4i_p;
#else
#define HAS_ASYNC_LDS 0
#endif

static __device__ __forceinline__ v16h cat16(v8h lo, v8h hi) {
    v16h r;
#pragma unroll
    for (int i = 0; i < 8; ++i) { r[i] = lo[i]; r[i + 8] = hi[i]; }
    return r;
}
// A fragment: lane holds K = kb..kb+15 (16B-aligned LDS loads)
static __device__ __forceinline__ v16h load_afrag(const _Float16* p) {
    return cat16(*(const v8h*)p, *(const v8h*)(p + 8));
}
// B fragment (K-transposed weights in LDS): 16 contiguous halfs
static __device__ __forceinline__ v16h load_bfrag(const _Float16* p) {
    return cat16(*(const v8h*)p, *(const v8h*)(p + 8));
}
static __device__ __forceinline__ v8f wmma_f16(v16h a, v16h b, v8f c) {
    return __builtin_amdgcn_wmma_f32_16x16x32_f16(false, a, false, b, (short)0, c, false, false);
}

// ---------------- kernel 0: weight conversion / pre-transpose -----------
__global__ __launch_bounds__(256) void convert_weights(
    const float* __restrict__ aw1, const float* __restrict__ fw1,
    const float* __restrict__ fw2,
    _Float16* __restrict__ aw1t,   // [64][HS]: aw1_bot K-transposed LDS image
    _Float16* __restrict__ fw1h, _Float16* __restrict__ fw2h) {
    int i = blockIdx.x * 256 + threadIdx.x;
    if (i < 64 * 64) {
        int n = i >> 6, k = i & 63;
        aw1t[n * HS + k] = (_Float16)aw1[(64 + k) * 64 + n];
    }
    if (i < 128 * 64) fw1h[i] = (_Float16)fw1[i];
    if (i < 64 * 32)  fw2h[i] = (_Float16)fw2[i];
}

// ---------------- kernel 1: fused attention (one block per batch row) ---
__global__ __launch_bounds__(256) void attn_fused(
    const float* __restrict__ target, const float* __restrict__ history,
    const float* __restrict__ aw1, const float* __restrict__ ab1,
    const float* __restrict__ aw2, const float* __restrict__ ab2,
    const _Float16* __restrict__ aw1t,
    float* __restrict__ xbuf, float* __restrict__ attn_out) {
    __shared__ _Float16 sHist[LPAD * HS];   // history tile, f16
    __shared__ _Float16 sWt[64 * HS];       // aw1_bot, K-transposed: [n][k]
    __shared__ float sTq[64];
    __shared__ float sAw2[64];
    __shared__ float sScores[LPAD];
    __shared__ float red[256];
    __shared__ float wred[4 * 64];

    const int b   = blockIdx.x;
    const int tid = threadIdx.x;

    // stage aw1_bot LDS image: pure byte copy (pre-transposed in workspace)
#if HAS_ASYNC_LDS
    {
        unsigned lbase = (unsigned)(unsigned long long)(&sWt[0]);
        for (int idx = tid; idx < 64 * HS / 8; idx += 256) {  // 16B chunks
            gv4i_p g = (gv4i_p)(unsigned long long)(aw1t + idx * 8);
            lv4i_p l = (lv4i_p)(unsigned long long)(lbase + idx * 16u);
            __builtin_amdgcn_global_load_async_to_lds_b128(g, l, 0, 0);
        }
    }
#else
    for (int idx = tid; idx < 64 * HS / 4; idx += 256)
        ((v4h*)sWt)[idx] = ((const v4h*)aw1t)[idx];
#endif

    // stage history -> f16 LDS, vectorized b128 loads (streaming NT)
    const v4f* hv = (const v4f*)(history + (size_t)b * LSEQ * DDIM);
    for (int idx = tid; idx < LPAD * DDIM / 4; idx += 256) {
        int r = idx >> 4, c4 = idx & 15;   // 16 float4 per 64-wide row
        v4f v = {0.0f, 0.0f, 0.0f, 0.0f};
        if (r < LSEQ) v = __builtin_nontemporal_load(hv + r * 16 + c4);
        v4h h;
#pragma unroll
        for (int j = 0; j < 4; ++j) h[j] = (_Float16)v[j];
        *(v4h*)(&sHist[r * HS + c4 * 4]) = h;
    }
    // tq[n] = ab1[n] + target[b] . aw1_top[:,n]   (tiny, L2 resident)
    if (tid < 64) {
        const float* trow = target + (size_t)b * 64;
        float acc = ab1[tid];
        for (int k = 0; k < 64; ++k) acc += trow[k] * aw1[k * 64 + tid];
        sTq[tid]  = acc;
        sAw2[tid] = aw2[tid];
    }
#if HAS_ASYNC_LDS
    __builtin_amdgcn_s_wait_asynccnt(0);
#endif
    __syncthreads();

    const int wave = tid >> 5;
    const int lane = tid & 31;
    const int mn   = lane & 15;
    const int hi   = lane >> 4;       // lane half select
    const int kbA  = hi * 16;         // A: K offset per lane-half
    const int koB  = hi * 16;         // B: K offset per lane-half
    const float b2 = ab2[0];

    // hoist all B fragments + epilogue scalars out of the tile loop
    v16h bfr[4][2];
    float tqv[4], w2v[4];
#pragma unroll
    for (int nt = 0; nt < 4; ++nt) {
        const _Float16* wb = &sWt[(nt * 16 + mn) * HS + koB];
        bfr[nt][0] = load_bfrag(wb);
        bfr[nt][1] = load_bfrag(wb + 32);
        tqv[nt] = sTq[nt * 16 + mn];
        w2v[nt] = sAw2[nt * 16 + mn];
    }

    for (int t = wave; t < LPAD / 16; t += 8) {   // uniform per wave
        const int row0 = t * 16;
        v16h a0 = load_afrag(&sHist[(row0 + mn) * HS + 0  + kbA]);
        v16h a1 = load_afrag(&sHist[(row0 + mn) * HS + 32 + kbA]);
        float part[8];
#pragma unroll
        for (int r = 0; r < 8; ++r) part[r] = 0.0f;
#pragma unroll
        for (int nt = 0; nt < 4; ++nt) {
            v8f c = {};
            c = wmma_f16(a0, bfr[nt][0], c);
            c = wmma_f16(a1, bfr[nt][1], c);
#pragma unroll
            for (int r = 0; r < 8; ++r) {
                float h = fmaxf(c[r] + tqv[nt], 0.0f);
                part[r] += h * w2v[nt];
            }
        }
        // reduce over the 16 lanes holding columns of the same row
#pragma unroll
        for (int r = 0; r < 8; ++r) {
            float v = part[r];
            for (int o = 1; o < 16; o <<= 1) v += __shfl_xor(v, o, 32);
            part[r] = v;
        }
        if (mn == 0) {   // lanes 0 (M=0..7) and 16 (M=8..15)
#pragma unroll
            for (int r = 0; r < 8; ++r)
                sScores[row0 + hi * 8 + r] = part[r] + b2;
        }
    }
    __syncthreads();

    // softmax over L=200
    float myv = (tid < LSEQ) ? sScores[tid] : -3.0e38f;
    red[tid] = myv;
    __syncthreads();
    for (int s = 128; s > 0; s >>= 1) {
        if (tid < s) red[tid] = fmaxf(red[tid], red[tid + s]);
        __syncthreads();
    }
    float mx = red[0];
    __syncthreads();
    float e = (tid < LSEQ) ? __expf(myv - mx) : 0.0f;
    red[tid] = e;
    __syncthreads();
    for (int s = 128; s > 0; s >>= 1) {
        if (tid < s) red[tid] += red[tid + s];
        __syncthreads();
    }
    float w = e * (1.0f / red[0]);
    __syncthreads();
    if (tid < LSEQ)
        __builtin_nontemporal_store(w, attn_out + (size_t)b * LSEQ + tid);
    if (tid < LPAD) sScores[tid] = (tid < LSEQ) ? w : 0.0f;
    __syncthreads();

    // weighted = sum_l w[l] * history[l,:]   (reuses LDS history, no HBM re-read)
    {
        int d = tid & 63, g = tid >> 6;
        float acc = 0.0f;
        for (int l = g; l < LSEQ; l += 4) acc += sScores[l] * (float)sHist[l * HS + d];
        wred[g * 64 + d] = acc;
        __syncthreads();
        if (g == 0) {
            float s = wred[d] + wred[64 + d] + wred[128 + d] + wred[192 + d];
            xbuf[(size_t)b * 128 + 64 + d] = s;
            xbuf[(size_t)b * 128 + d]      = target[(size_t)b * 64 + d];
        }
    }
}

// -------- column stats, stage 1: coalesced row-band partials ------------
template <int C>
__global__ __launch_bounds__(256) void colstats_partial(
    const float* __restrict__ X, int bandRows,
    float* __restrict__ psum, float* __restrict__ psumsq) {
    constexpr int RPB = 256 / C;     // row-groups per block (2/4/8)
    __shared__ float s1[256], s2[256];
    const int t = threadIdx.x;
    const int c = t % C, rg = t / C;
    const int r0 = blockIdx.x * bandRows;
    float a = 0.0f, q = 0.0f;
    for (int r = rg; r < bandRows; r += RPB) {
        float v = X[(size_t)(r0 + r) * C + c];   // consecutive tids -> consecutive cols
        a += v; q += v * v;
    }
    s1[t] = a; s2[t] = q;
    __syncthreads();
    for (int s = 128; s >= C; s >>= 1) {
        if (t < s) { s1[t] += s1[t + s]; s2[t] += s2[t + s]; }
        __syncthreads();
    }
    if (t < C) {
        psum[blockIdx.x * C + t]   = s1[t];
        psumsq[blockIdx.x * C + t] = s2[t];
    }
}

// -------- column stats, stage 2: finalize mean / rsqrt(var+eps) ---------
__global__ __launch_bounds__(64) void colstats_final(
    const float* __restrict__ psum, const float* __restrict__ psumsq,
    int nparts, int C, int Bn,
    float* __restrict__ mean, float* __restrict__ rstd) {
    int c = blockIdx.x * 64 + threadIdx.x;
    if (c >= C) return;
    float a = 0.0f, q = 0.0f;
    for (int p = 0; p < nparts; ++p) { a += psum[p * C + c]; q += psumsq[p * C + c]; }
    float m = a / (float)Bn;
    float v = q / (float)Bn - m * m;
    mean[c] = m;
    rstd[c] = rsqrtf(v + BN_EPS);
}

// -------- fused BN -> GEMM (WMMA) -> bias -> ReLU ; block = 128 rows ----
template <int K, int N>
__global__ __launch_bounds__(256) void fc_bn_relu(
    const float* __restrict__ X, const _Float16* __restrict__ Wh,
    const float* __restrict__ bias,
    const float* __restrict__ mean, const float* __restrict__ rstd,
    const float* __restrict__ gamma, const float* __restrict__ beta,
    float* __restrict__ Y) {
    constexpr int KS = K + 8;                 // padded stride (bytes mult of 16)
    constexpr int KV = K / 4;
    __shared__ _Float16 sA[128 * KS];
    __shared__ _Float16 sWt[N * KS];          // [n][k]
    const int tid = threadIdx.x;
    const int r0  = blockIdx.x * 128;

    for (int idx = tid; idx < 128 * KV; idx += 256) {
        int r = idx / KV, c4 = idx % KV;
        v4f x = *(const v4f*)(X + (size_t)(r0 + r) * K + c4 * 4);
        v4f m = *(const v4f*)(mean  + c4 * 4);
        v4f s = *(const v4f*)(rstd  + c4 * 4);
        v4f g = *(const v4f*)(gamma + c4 * 4);
        v4f o = *(const v4f*)(beta  + c4 * 4);
        v4h h;
#pragma unroll
        for (int j = 0; j < 4; ++j) h[j] = (_Float16)((x[j] - m[j]) * s[j] * g[j] + o[j]);
        *(v4h*)(&sA[r * KS + c4 * 4]) = h;
    }
    for (int idx = tid; idx < K * N / 4; idx += 256) {
        int k = idx / (N / 4), n4 = idx % (N / 4);
        v4h w = *(const v4h*)(Wh + k * N + n4 * 4);
#pragma unroll
        for (int j = 0; j < 4; ++j) sWt[(n4 * 4 + j) * KS + k] = w[j];
    }
    __syncthreads();

    const int wave = tid >> 5, lane = tid & 31;
    const int mn = lane & 15, hi = lane >> 4;
    const int kbA = hi * 16, koB = hi * 16;
    const int row0 = wave * 16;
    constexpr int KSTEPS = K / 32;
    constexpr int NT = N / 16;

    v16h a[KSTEPS];
#pragma unroll
    for (int s = 0; s < KSTEPS; ++s)
        a[s] = load_afrag(&sA[(row0 + mn) * KS + 32 * s + kbA]);

#pragma unroll
    for (int nt = 0; nt < NT; ++nt) {
        v8f c = {};
#pragma unroll
        for (int s = 0; s < KSTEPS; ++s) {
            v16h bb = load_bfrag(&sWt[(nt * 16 + mn) * KS + 32 * s + koB]);
            c = wmma_f16(a[s], bb, c);
        }
        float bs = bias[nt * 16 + mn];
#pragma unroll
        for (int r = 0; r < 8; ++r) {
            float v = fmaxf(c[r] + bs, 0.0f);
            int row = r0 + row0 + hi * 8 + r;
            Y[(size_t)row * N + nt * 16 + mn] = v;
        }
    }
}

// -------- final layer: BN(32) -> dot(32) -> logit -----------------------
__global__ __launch_bounds__(256) void fc3_out(
    const float* __restrict__ X, const float* __restrict__ fw3,
    const float* __restrict__ fb3,
    const float* __restrict__ mean, const float* __restrict__ rstd,
    const float* __restrict__ gamma, const float* __restrict__ beta,
    float* __restrict__ out) {
    int b = blockIdx.x * 256 + threadIdx.x;
    if (b >= BATCH) return;
    float acc = fb3[0];
#pragma unroll
    for (int c = 0; c < 32; ++c) {
        float v = (X[(size_t)b * 32 + c] - mean[c]) * rstd[c] * gamma[c] + beta[c];
        acc += v * fw3[c];
    }
    out[b] = acc;
}

extern "C" void kernel_launch(void* const* d_in, const int* in_sizes, int n_in,
                              void* d_out, int out_size, void* d_ws, size_t ws_size,
                              hipStream_t stream) {
    (void)in_sizes; (void)n_in; (void)out_size; (void)ws_size;
    const float* target = (const float*)d_in[0];
    const float* history = (const float*)d_in[1];
    const float* aw1 = (const float*)d_in[2];
    const float* ab1 = (const float*)d_in[3];
    const float* aw2 = (const float*)d_in[4];
    const float* ab2 = (const float*)d_in[5];
    const float* bn0_g = (const float*)d_in[6];
    const float* bn0_b = (const float*)d_in[7];
    const float* fw1 = (const float*)d_in[8];
    const float* fb1 = (const float*)d_in[9];
    const float* bn1_g = (const float*)d_in[10];
    const float* bn1_b = (const float*)d_in[11];
    const float* fw2 = (const float*)d_in[12];
    const float* fb2 = (const float*)d_in[13];
    const float* bn2_g = (const float*)d_in[14];
    const float* bn2_b = (const float*)d_in[15];
    const float* fw3 = (const float*)d_in[16];
    const float* fb3 = (const float*)d_in[17];

    float* logits = (float*)d_out;                   // [B]
    float* attnw  = (float*)d_out + BATCH;           // [B, L]

    char* ws = (char*)d_ws;
    size_t o = 0;
    _Float16* aw1t = (_Float16*)(ws + o); o += ((size_t)64 * HS * 2 + 255) & ~255ull;
    _Float16* fw1h = (_Float16*)(ws + o); o += ((size_t)128 * 64 * 2 + 255) & ~255ull;
    _Float16* fw2h = (_Float16*)(ws + o); o += ((size_t)64 * 32 * 2 + 255) & ~255ull;
    float* xbuf  = (float*)(ws + o); o += (size_t)BATCH * 128 * 4;
    float* x1buf = (float*)(ws + o); o += (size_t)BATCH * 64 * 4;
    float* x2buf = (float*)(ws + o); o += (size_t)BATCH * 32 * 4;
    float* stats = (float*)(ws + o); o += 2048;
    float* psum   = (float*)(ws + o); o += (size_t)32 * 128 * 4;
    float* psumsq = (float*)(ws + o); o += (size_t)32 * 128 * 4;
    float* mean0 = stats;          float* rstd0 = stats + 128;
    float* mean1 = stats + 256;    float* rstd1 = stats + 320;
    float* mean2 = stats + 384;    float* rstd2 = stats + 416;

    convert_weights<<<32, 256, 0, stream>>>(aw1, fw1, fw2, aw1t, fw1h, fw2h);

    attn_fused<<<BATCH, 256, 0, stream>>>(target, history, aw1, ab1, aw2, ab2,
                                          aw1t, xbuf, attnw);

    // BN0 stats -> fc1
    colstats_partial<128><<<32, 256, 0, stream>>>(xbuf, BATCH / 32, psum, psumsq);
    colstats_final<<<2, 64, 0, stream>>>(psum, psumsq, 32, 128, BATCH, mean0, rstd0);
    fc_bn_relu<128, 64><<<BATCH / 128, 256, 0, stream>>>(
        xbuf, fw1h, fb1, mean0, rstd0, bn0_g, bn0_b, x1buf);

    // BN1 stats -> fc2
    colstats_partial<64><<<32, 256, 0, stream>>>(x1buf, BATCH / 32, psum, psumsq);
    colstats_final<<<1, 64, 0, stream>>>(psum, psumsq, 32, 64, BATCH, mean1, rstd1);
    fc_bn_relu<64, 32><<<BATCH / 128, 256, 0, stream>>>(
        x1buf, fw2h, fb2, mean1, rstd1, bn1_g, bn1_b, x2buf);

    // BN2 stats -> logits
    colstats_partial<32><<<32, 256, 0, stream>>>(x2buf, BATCH / 32, psum, psumsq);
    colstats_final<<<1, 64, 0, stream>>>(psum, psumsq, 32, 32, BATCH, mean2, rstd2);
    fc3_out<<<BATCH / 256, 256, 0, stream>>>(x2buf, fw3, fb3, mean2, rstd2,
                                             bn2_g, bn2_b, logits);
}
